// MultiScaleResidualVectorQuantize_13030930776316
// MI455X (gfx1250) — matrix operand order repaired and checked by reading, not compiled, not run on hardware
//
#include <hip/hip_runtime.h>
#include <hip/hip_bf16.h>

typedef __attribute__((ext_vector_type(16))) _Float16 v16h;
typedef __attribute__((ext_vector_type(8)))  float    v8f;
typedef __attribute__((ext_vector_type(4))) unsigned int u32x4;
typedef __attribute__((ext_vector_type(8))) int i32x8;
typedef __attribute__((ext_vector_type(4))) int i32x4;

#define WMMA_F16(A,B,C) __builtin_amdgcn_wmma_f32_16x16x32_f16(false,(A),false,(B),(short)0,(C),false,false)

#if __has_builtin(__builtin_amdgcn_tensor_load_to_lds) && __has_builtin(__builtin_amdgcn_s_wait_tensorcnt)
#define USE_TDM 1
#else
#define USE_TDM 0
#endif

// ---- problem constants ----
// z: (16,512,4096) f32, in_w (4,8,512), in_b (4,8), out_w (4,512,8), out_b (4,512), codebooks (4,1024,8)
// outputs flat f32: z_q | latents | commit | cbloss | codes0..3
static constexpr size_t kO_zq     = 0ULL;
static constexpr size_t kO_lat    = 33554432ULL;            // 16*512*4096
static constexpr size_t kO_commit = 35651584ULL;            // + 16*32*4096
static constexpr size_t kO_cbloss = 35651585ULL;
static constexpr size_t kO_codes  = 35651586ULL;            // codes st base = kO_codes + 8192*((1<<st)-1)

// ---- workspace offsets (bytes) ----
static constexpr size_t kW_cbn  = 0;        // 4*1024*8 f16  (row-normalized codebooks)
static constexpr size_t kW_W    = 65536;    // 32*512  f16   (stacked in_w)
static constexpr size_t kW_OutW = 98304;    // 512*32  f16   (stacked out_w, [d][j*8+c])
static constexpr size_t kW_M    = 131072;   // 4*4*8*8 f32   (M[i][j] = in_w_i . out_w_j)
static constexpr size_t kW_v    = 135168;   // 4*4*8   f32   (v[i][j] = in_w_i . out_b_j)
static constexpr size_t kW_bsum = 135680;   // 512     f32   (sum_j out_b_j)

__device__ __forceinline__ int qoff(int j) { return 16 * ((1 << j) - 1); } // 0,16,48,112

// ============================= prep kernel =============================
__global__ void msrvq_prep(const float* __restrict__ in_w, const float* __restrict__ in_b,
                           const float* __restrict__ out_w, const float* __restrict__ out_b,
                           const float* __restrict__ cb,
                           _Float16* __restrict__ cbn_h, _Float16* __restrict__ W_h,
                           _Float16* __restrict__ OutW_h, float* __restrict__ Mcorr,
                           float* __restrict__ vcorr, float* __restrict__ biasSum,
                           float* __restrict__ out) {
  int gid = blockIdx.x * blockDim.x + threadIdx.x;
  int nthr = gridDim.x * blockDim.x;
  // normalized codebooks -> f16
  for (int code = gid; code < 4096; code += nthr) {
    float v[8], ss = 0.f;
    for (int c = 0; c < 8; c++) { v[c] = cb[code * 8 + c]; ss += v[c] * v[c]; }
    float inv = 1.0f / fmaxf(sqrtf(ss), 1e-12f);
    for (int c = 0; c < 8; c++) cbn_h[code * 8 + c] = (_Float16)(v[c] * inv);
  }
  // stacked in_w (already [32][512] when flattened)
  for (int i = gid; i < 32 * 512; i += nthr) W_h[i] = (_Float16)in_w[i];
  // stacked out_w: OutW[d][j*8+c] = out_w[j][d][c]
  for (int i = gid; i < 512 * 32; i += nthr) {
    int d = i >> 5, k = i & 31, j = k >> 3, c = k & 7;
    OutW_h[i] = (_Float16)out_w[(j * 512 + d) * 8 + c];
  }
  // M[i][j][c][c2] = sum_d in_w[i][c][d] * out_w[j][d][c2]
  for (int idx = gid; idx < 1024; idx += nthr) {
    int c2 = idx & 7, c = (idx >> 3) & 7, j = (idx >> 6) & 3, i = (idx >> 8) & 3;
    float acc = 0.f;
    for (int d = 0; d < 512; d++) acc += in_w[(i * 8 + c) * 512 + d] * out_w[(j * 512 + d) * 8 + c2];
    Mcorr[idx] = acc;
  }
  // v[i][j][c] = sum_d in_w[i][c][d] * out_b[j][d]
  for (int idx = gid; idx < 128; idx += nthr) {
    int c = idx & 7, j = (idx >> 3) & 3, i = (idx >> 5) & 3;
    float acc = 0.f;
    for (int d = 0; d < 512; d++) acc += in_w[(i * 8 + c) * 512 + d] * out_b[j * 512 + d];
    vcorr[idx] = acc;
  }
  for (int d = gid; d < 512; d += nthr) {
    float a = 0.f;
    for (int j = 0; j < 4; j++) a += out_b[j * 512 + d];
    biasSum[d] = a;
  }
  if (gid == 0) { out[kO_commit] = 0.f; out[kO_cbloss] = 0.f; }
}

// ============================= main fused kernel =============================
// grid (32 chunks, 16 batches), 256 threads = 8 wave32s; chunk = 128 timesteps
__global__ __launch_bounds__(256)
void msrvq_main(const float* __restrict__ z, const float* __restrict__ in_b,
                const float* __restrict__ cb,
                const _Float16* __restrict__ cbn_h, const _Float16* __restrict__ W_h,
                const _Float16* __restrict__ OutW_h, const float* __restrict__ Mcorr,
                const float* __restrict__ vcorr, const float* __restrict__ biasSum,
                float* __restrict__ out) {
  __shared__ __align__(16) _Float16 sCB[1024 * 8];   // one stage's normalized codebook (16KB)
  __shared__ float sY[32][132];                      // in-proj of z for all 4 stages (padded)
  __shared__ float sE[128][8];                       // per-stage encodings
  __shared__ float sQ[240][8];                       // selected codes, all stages (qoff)
  __shared__ float sM[4][4][8][8];
  __shared__ float sV[4][4][8];
  __shared__ float sB[32];
  __shared__ float sCandV[8][16];
  __shared__ int   sCandI[8][16];
  __shared__ float sLoss;

  const int tid  = threadIdx.x;
  const int lane = tid & 31;
  const int wave = tid >> 5;
  const int g    = lane >> 4;   // C/B fragment half
  const int ln   = lane & 15;
  const int b    = blockIdx.y;
  const int chunk = blockIdx.x;
  const int t0   = chunk * 128;
  const size_t zbase = (size_t)b * 512 * 4096;

  // init small LDS tables
  for (int i = tid; i < 1024; i += 256) ((float*)sM)[i] = Mcorr[i];
  for (int i = tid; i < 128; i += 256) ((float*)sV)[i] = vcorr[i];
  for (int i = tid; i < 32; i += 256) sB[i] = in_b[i];
  if (tid == 0) sLoss = 0.f;

  // ---------- Phase A: Y = W_all(32x512) @ z_chunk(512x128) via WMMA ----------
  {
    const int nt = wave;                    // N-tile = 16 timesteps
    const int t = t0 + nt * 16 + ln;
    v8f acc0 = {}, acc1 = {};
    for (int k = 0; k < 16; k++) {          // K steps of 32 channels
      if (k < 15) __builtin_prefetch(&z[zbase + (size_t)((k + 1) * 32 + g * 16) * 4096 + t], 0, 0);
      v16h bfr;
      #pragma unroll
      for (int h = 0; h < 16; h++) {        // B: K = g*16 + h, col = ln
        int d = k * 32 + g * 16 + h;
        bfr[h] = (_Float16)z[zbase + (size_t)d * 4096 + t];
      }
      v16h a0, a1;
      #pragma unroll
      for (int h = 0; h < 16; h++) {        // A: row = ln, K per half/VGPR map
        int K = g * 16 + h;
        a0[h] = W_h[(ln)      * 512 + k * 32 + K];
        a1[h] = W_h[(16 + ln) * 512 + k * 32 + K];
      }
      acc0 = WMMA_F16(a0, bfr, acc0);
      acc1 = WMMA_F16(a1, bfr, acc1);
    }
    #pragma unroll
    for (int r = 0; r < 8; r++) {           // C: row = g*8 + r, col = ln
      sY[g * 8 + r][nt * 16 + ln]      = acc0[r];
      sY[16 + g * 8 + r][nt * 16 + ln] = acc1[r];
    }
  }
  __syncthreads();

  float lossAcc = 0.f;

  // ---------- Phase B: 4-stage cascade ----------
  for (int st = 0; st < 4; st++) {
    const int stride = 8 >> st;
    const int nTok = 16 << st;

#if USE_TDM
    // stage codebook -> LDS via Tensor Data Mover (async; overlapped with E compute)
    if (wave == 0) {
      unsigned long long ga = (unsigned long long)(const void*)(cbn_h + (size_t)st * 1024 * 8);
      unsigned int ldsAddr = (unsigned int)(unsigned long long)(uintptr_t)(void*)sCB;
      // D# group0: count=1 | lds_addr | global_addr | type=2
      u32x4 g0;
      g0[0] = 1u;                                   // count=1, user mode
      g0[1] = ldsAddr;                              // LDS byte address
      g0[2] = (unsigned int)(ga & 0xffffffffull);   // global_addr[31:0]
      g0[3] = (unsigned int)((ga >> 32) & 0x1ffffffull) | (2u << 30); // addr[56:32] | type=2
      // D# group1: data_size=8B; tensor_dim0=tile_dim0=2048; dim1=1; stride0=2048
      i32x8 g1;
      g1[0] = 0x00030000;   // workgroup_mask=0, data_size=3 (8B)
      g1[1] = (int)(2048u << 16);   // tensor_dim0[15:0] << 16 (bits 63:48)
      g1[2] = (int)(1u << 16);      // tensor_dim0 hi=0 | tensor_dim1 lo=1 (bits 95:80)
      g1[3] = (int)(2048u << 16);   // tensor_dim1 hi=0 | tile_dim0=2048 (bits 127:112)
      g1[4] = 1;                    // tile_dim1=1, tile_dim2=0
      g1[5] = 2048;                 // tensor_dim0_stride lo
      g1[6] = (int)(2048u << 16);   // stride0 hi=0 | tensor_dim1_stride lo16=2048
      g1[7] = 0;
      i32x4 gz = {};                // groups 2/3: unused (<=2D)
#if __clang_major__ >= 23
      i32x8 gz8 = {};
      __builtin_amdgcn_tensor_load_to_lds(g0, g1, gz, gz, gz8, 0);
#else
      __builtin_amdgcn_tensor_load_to_lds(g0, g1, gz, gz, 0);
#endif
    }
#else
    {
      const uint4* src = (const uint4*)(cbn_h + (size_t)st * 1024 * 8);
      uint4* dst = (uint4*)sCB;
      for (int i = tid; i < 1024; i += 256) dst[i] = src[i];
    }
#endif

    // e = pool(Y rows) + b - sum_{j<st} (M@q_j + v)   (runs while TDM streams the codebook)
    for (int i = tid; i < nTok * 8; i += 256) {
      int u = i >> 3, c = i & 7;
      float acc = 0.f;
      for (int tt = 0; tt < stride; tt++) acc += sY[st * 8 + c][u * stride + tt];
      acc = acc * (1.0f / (float)stride) + sB[st * 8 + c];
      for (int j = 0; j < st; j++) {
        int uj = u >> (st - j);
        const float* q = &sQ[qoff(j) + uj][0];
        float dot = sV[st][j][c];
        for (int c2 = 0; c2 < 8; c2++) dot += sM[st][j][c][c2] * q[c2];
        acc -= dot;
      }
      sE[u][c] = acc;
    }
#if USE_TDM
    if (wave == 0) __builtin_amdgcn_s_wait_tensorcnt(0);  // codebook DMA complete
#endif
    __syncthreads();

    // latents: repeat(e, stride) -> out
    for (int i = tid; i < 8 * 128; i += 256) {
      int c = i >> 7, tt = i & 127;
      out[kO_lat + ((size_t)(b * 32 + st * 8 + c)) * 4096 + t0 + tt] = sE[tt >> (3 - st)][c];
    }

    // ---- WMMA nearest-neighbor search: scores = E(16x8) @ cbn^T(8x1024) ----
    const int nGroup = nTok >> 4;         // 1,2,4,8 token groups of 16
    const int wpg = 8 / nGroup;           // waves per group
    const int grp = wave / wpg;
    const int ww = wave % wpg;
    const int tilesPerWave = 64 / wpg;    // code tiles (16 codes each)

    v16h af = {};
    if (lane < 16) {                      // A: lanes 0-15 hold K=0..7 (rest zero-padded)
      #pragma unroll
      for (int c = 0; c < 8; c++) af[c] = (_Float16)sE[grp * 16 + ln][c];
    }
    float bv[8]; int bi[8];
    #pragma unroll
    for (int r = 0; r < 8; r++) { bv[r] = -3.4e38f; bi[r] = 0; }

    for (int ti = 0; ti < tilesPerWave; ti++) {
      int ct = ww * tilesPerWave + ti;
      v16h bf = {};
      if (lane < 16) {                    // B: lanes 0-15 hold K=0..7 = cbn row (rest zero)
        #pragma unroll
        for (int c = 0; c < 8; c++) bf[c] = sCB[(ct * 16 + ln) * 8 + c];
      }
      v8f cc = {};
      cc = WMMA_F16(af, bf, cc);
      #pragma unroll
      for (int r = 0; r < 8; r++) {       // running per-lane argmax over this lane's column
        float sc = cc[r];
        int id = ct * 16 + ln;
        if (sc > bv[r] || (sc == bv[r] && id < bi[r])) { bv[r] = sc; bi[r] = id; }
      }
    }
    // reduce across the 16 lanes of each half (rows g*8+r live in lanes g*16..g*16+15)
    #pragma unroll
    for (int r = 0; r < 8; r++) {
      #pragma unroll
      for (int m = 1; m < 16; m <<= 1) {
        float ov = __shfl_xor(bv[r], m);
        int   oi = __shfl_xor(bi[r], m);
        if (ov > bv[r] || (ov == bv[r] && oi < bi[r])) { bv[r] = ov; bi[r] = oi; }
      }
    }
    if (ln == 0) {
      #pragma unroll
      for (int r = 0; r < 8; r++) { sCandV[wave][g * 8 + r] = bv[r]; sCandI[wave][g * 8 + r] = bi[r]; }
    }
    __syncthreads();

    // combine across waves, fetch q, losses, codes
    for (int tk = tid; tk < nTok; tk += 256) {
      int grp2 = tk >> 4, tl = tk & 15;
      float best = -3.4e38f; int bidx = 0;
      for (int w2 = grp2 * wpg; w2 < grp2 * wpg + wpg; w2++) {
        float v = sCandV[w2][tl]; int ii = sCandI[w2][tl];
        if (v > best || (v == best && ii < bidx)) { best = v; bidx = ii; }
      }
      // code index output (as float, matching flat f32 output buffer)
      size_t cbase = kO_codes + (size_t)8192 * ((1u << st) - 1u);
      out[cbase + (size_t)b * (512 << st) + (size_t)chunk * (16 << st) + tk] = (float)bidx;
      // unnormalized code vector -> sQ; loss accumulation
      const float* q = cb + ((size_t)(st * 1024 + bidx)) * 8;
      float ls = 0.f;
      for (int c = 0; c < 8; c++) {
        float qv = q[c];
        float dd = sE[tk][c] - qv;
        ls += dd * dd;
        sQ[qoff(st) + tk][c] = qv;
      }
      lossAcc += ls * (1.0f / (128.0f * (float)(512 << st)));  // 1/(B*C*Ts)
    }
    __syncthreads();
  }

  // ---------- Phase C: z_q = OutW_all(512x32) @ Qstack(32x128) + bias ----------
  {
    const int nt = wave;
    v16h bf;
    #pragma unroll
    for (int h = 0; h < 16; h++) {        // B: K = g*16+h -> (stage j, comp c); col = timestep
      int K = g * 16 + h;
      int j = K >> 3, c = K & 7;
      int tc = nt * 16 + ln;
      bf[h] = (_Float16)sQ[qoff(j) + (tc >> (3 - j))][c];
    }
    for (int mt = 0; mt < 32; mt++) {
      v16h a;
      #pragma unroll
      for (int h = 0; h < 16; h++) {
        int K = g * 16 + h;
        a[h] = OutW_h[(mt * 16 + ln) * 32 + K];
      }
      v8f cc = {};
      cc = WMMA_F16(a, bf, cc);
      #pragma unroll
      for (int r = 0; r < 8; r++) {
        int d = mt * 16 + g * 8 + r;
        out[kO_zq + zbase + (size_t)d * 4096 + t0 + nt * 16 + ln] = cc[r] + biasSum[d];
      }
    }
  }

  // losses (commit == cbloss in forward)
  atomicAdd(&sLoss, lossAcc);
  __syncthreads();
  if (tid == 0) {
    atomicAdd(&out[kO_commit], sLoss);
    atomicAdd(&out[kO_cbloss], sLoss);
  }
}

// ============================= launch =============================
extern "C" void kernel_launch(void* const* d_in, const int* in_sizes, int n_in,
                              void* d_out, int out_size, void* d_ws, size_t ws_size,
                              hipStream_t stream) {
  const float* z     = (const float*)d_in[0];
  const float* in_w  = (const float*)d_in[1];
  const float* in_b  = (const float*)d_in[2];
  const float* out_w = (const float*)d_in[3];
  const float* out_b = (const float*)d_in[4];
  const float* cb    = (const float*)d_in[5];
  float* out = (float*)d_out;
  char* ws = (char*)d_ws;

  _Float16* cbn_h  = (_Float16*)(ws + kW_cbn);
  _Float16* W_h    = (_Float16*)(ws + kW_W);
  _Float16* OutW_h = (_Float16*)(ws + kW_OutW);
  float* Mcorr   = (float*)(ws + kW_M);
  float* vcorr   = (float*)(ws + kW_v);
  float* biasSum = (float*)(ws + kW_bsum);

  msrvq_prep<<<dim3(64), dim3(256), 0, stream>>>(in_w, in_b, out_w, out_b, cb,
                                                 cbn_h, W_h, OutW_h, Mcorr, vcorr, biasSum, out);
  msrvq_main<<<dim3(32, 16), dim3(256), 0, stream>>>(z, in_b, cb, cbn_h, W_h, OutW_h,
                                                     Mcorr, vcorr, biasSum, out);
}
